// CompletionLoss_27221502722180
// MI455X (gfx1250) — matrix operand order repaired and checked by reading, not compiled, not run on hardware
//
#include <hip/hip_runtime.h>
#include <hip/hip_bf16.h>
#include <stdint.h>

#define T_DIM 4096
#define D_DIM 1024
#define SENTINEL_F 9999.0f
#define ROW_PENALTY 10.0f

typedef __attribute__((ext_vector_type(16))) __bf16 v16bf;
typedef __attribute__((ext_vector_type(8)))  float  v8f;
typedef __attribute__((ext_vector_type(4)))  unsigned int v4u;
typedef __attribute__((ext_vector_type(8)))  int v8i_t;
typedef __attribute__((ext_vector_type(4)))  int v4i_t;

#define WMMA_BF16(acc, a, b)                                                  \
  acc = __builtin_amdgcn_wmma_f32_16x16x32_bf16(false, (a), false, (b),       \
                                                (short)0, (acc), false, false)

// ---- LDS tile geometry: 32-wide K slabs, 16B pad per 64B row (bank-conflict free)
#define KSTEP 32
#define ROW_BYTES 80                       // 64B data + 16B TDM pad
#define A_ROWS 64
#define B_ROWS 128
#define A_MAT_BYTES (A_ROWS * ROW_BYTES)   // 5120
#define B_MAT_BYTES (B_ROWS * ROW_BYTES)   // 10240
#define STAGE_BYTES (3 * A_MAT_BYTES + 3 * B_MAT_BYTES)  // 46080
#define NSTAGES (D_DIM / KSTEP)            // 32

__device__ __forceinline__ v16bf load16(const unsigned char* p0, const unsigned char* p1) {
  v16bf r;
  ((uint4*)&r)[0] = *((const uint4*)p0);
  ((uint4*)&r)[1] = *((const uint4*)p1);
  return r;
}

// ---- Tensor Data Mover: load a [rows x 32] bf16 tile (row stride D_DIM) into LDS,
//      padding 16B after every 64B row so LDS row stride becomes 80B.
__device__ __forceinline__ void tdm_load_tile(const __bf16* gbase, int row0, int k,
                                              unsigned int lds_addr, int rows) {
  const unsigned long long ga = (unsigned long long)(size_t)gbase +
      ((unsigned long long)(unsigned)row0 * D_DIM + (unsigned)k) * 2ull;
  v4u g0;
  g0[0] = 1u;                                                // count=1, user mode
  g0[1] = lds_addr;                                          // LDS byte address
  g0[2] = (unsigned int)ga;                                  // global_addr[31:0]
  g0[3] = (unsigned int)((ga >> 32) & 0x1FFFFFFull) | (2u << 30);  // [56:32] | type=2
  v8i_t g1;
  g1[0] = (int)((1u << 16)      // data_size = 1 -> 2 bytes
              | (1u << 20)      // pad_enable
              | (3u << 22)      // pad_interval: 16 DWORDs (64B) between pads
              | (3u << 25));    // pad_amount: 4 DWORDs (16B)
  g1[1] = (int)(((unsigned)D_DIM & 0xFFFFu) << 16);          // tensor_dim0[15:0]
  g1[2] = (int)(((unsigned)T_DIM & 0xFFFFu) << 16);          // tensor_dim1[15:0]
  g1[3] = (int)(32u << 16);                                  // tile_dim0 = 32
  g1[4] = rows;                                              // tile_dim1
  g1[5] = D_DIM;                                             // tensor_dim0_stride
  g1[6] = 0;
  g1[7] = 0;
  v4i_t z4 = {0, 0, 0, 0};
  v8i_t z8 = {0, 0, 0, 0, 0, 0, 0, 0};
  __builtin_amdgcn_tensor_load_to_lds(g0, g1, z4, z4, z8, 0);
}

__device__ __forceinline__ void tdm_stage(const __bf16* Pmb, const __bf16* Pmh,
                                          const __bf16* Pmh2, int ibase, int jbase,
                                          int k, unsigned int lds_stage) {
  tdm_load_tile(Pmb,  ibase, k, lds_stage + 0 * A_MAT_BYTES, A_ROWS);
  tdm_load_tile(Pmh,  ibase, k, lds_stage + 1 * A_MAT_BYTES, A_ROWS);
  tdm_load_tile(Pmh2, ibase, k, lds_stage + 2 * A_MAT_BYTES, A_ROWS);
  tdm_load_tile(Pmb,  jbase, k, lds_stage + 3 * A_MAT_BYTES + 0 * B_MAT_BYTES, B_ROWS);
  tdm_load_tile(Pmh,  jbase, k, lds_stage + 3 * A_MAT_BYTES + 1 * B_MAT_BYTES, B_ROWS);
  tdm_load_tile(Pmh2, jbase, k, lds_stage + 3 * A_MAT_BYTES + 2 * B_MAT_BYTES, B_ROWS);
}

// ---------------- K0: init per-row min keys ----------------
__global__ void k_init(unsigned long long* __restrict__ minkeys) {
  int i = blockIdx.x * blockDim.x + threadIdx.x;
  if (i < T_DIM) minkeys[i] = 0xFFFFFFFFFFFFFFFFull;
}

// ---------------- K1: pack bf16 operands, row argmin(M), row mse ----------------
__global__ __launch_bounds__(256) void k_pack(
    const float* __restrict__ X, const float* __restrict__ H,
    const float* __restrict__ C, const float* __restrict__ M,
    __bf16* __restrict__ Pmb, __bf16* __restrict__ Pmh, __bf16* __restrict__ Pmh2,
    int* __restrict__ am, float* __restrict__ row_mse) {
  const int row = blockIdx.x;
  const int tid = threadIdx.x;
  const size_t base = (size_t)row * D_DIM;

  float bestv = 3.4e38f;
  int bestd = 0;
  float acc = 0.0f;
  for (int d = tid; d < D_DIM; d += 256) {
    const size_t idx = base + d;
    const float m = M[idx];
    const float h = H[idx];
    const float mb = (m > 0.0f) ? 1.0f : 0.0f;
    const float mh = mb * h;
    Pmb[idx]  = (__bf16)mb;
    Pmh[idx]  = (__bf16)mh;
    Pmh2[idx] = (__bf16)(mh * h);
    if (m < bestv) { bestv = m; bestd = d; }  // first occurrence within stride
    const float r = (X[idx] - (h - C[idx])) * m;
    acc += r * r;
  }
  __shared__ float sv[256];
  __shared__ int   si[256];
  __shared__ float ss[256];
  sv[tid] = bestv; si[tid] = bestd; ss[tid] = acc;
  __syncthreads();
  for (int s = 128; s > 0; s >>= 1) {
    if (tid < s) {
      const float v2 = sv[tid + s];
      const int   d2 = si[tid + s];
      if (v2 < sv[tid] || (v2 == sv[tid] && d2 < si[tid])) { sv[tid] = v2; si[tid] = d2; }
      ss[tid] += ss[tid + s];
    }
    __syncthreads();
  }
  if (tid == 0) { am[row] = si[0]; row_mse[row] = ss[0]; }
}

// ---------------- K2: TDM-staged fused pairwise moment GEMMs + masked argmin ------
// Workgroup: 8 waves as 2x4 -> 64 rows x 128 cols; wave tile 32x32 (2x2 WMMA subtiles)
__global__ __launch_bounds__(256) void k_pair(
    const __bf16* __restrict__ Pmb, const __bf16* __restrict__ Pmh,
    const __bf16* __restrict__ Pmh2, const int* __restrict__ am,
    unsigned long long* __restrict__ minkeys) {
  __shared__ unsigned char smem[2][STAGE_BYTES];

  const int tid  = threadIdx.x;
  const int lane = tid & 31;
  const int wave = tid >> 5;
  const int wr = wave >> 2;                    // 0..1 : 32-row strip
  const int wc = wave & 3;                     // 0..3 : 32-col strip
  const int ibase = blockIdx.y * 64;
  const int jbase = blockIdx.x * 128;
  const int half = lane >> 4;
  const int l15  = lane & 15;

  const unsigned int ldsbase = (unsigned int)(size_t)(&smem[0][0]);

  v8f acc[2][2][6] = {};  // [asub][nsub][nn, ab, ba, cb, bc, dd]

  // Prologue: two stages in flight (wave 0 drives the TDM)
  if (tid < 32) {
    tdm_stage(Pmb, Pmh, Pmh2, ibase, jbase, 0,     ldsbase);
    tdm_stage(Pmb, Pmh, Pmh2, ibase, jbase, KSTEP, ldsbase + STAGE_BYTES);
  }

  // Per-lane LDS fragment offsets (ISA WMMA A/B layouts), 80B row stride
  const int rowA0 = wr * 32 + l15;           // asub 0 row
  const int aoff  = half * 16;               // bytes: K chunk {0..7} or {8..15}
  const int colB0 = wc * 32 + l15;           // nsub 0 col
  const int boff  = half * 16;               // bytes: K chunk {0..7} or {8..15}

  for (int ks = 0; ks < NSTAGES; ++ks) {
    const int p = ks & 1;
    if (tid < 32) {
      if (ks < NSTAGES - 1) __builtin_amdgcn_s_wait_tensorcnt(6);
      else                  __builtin_amdgcn_s_wait_tensorcnt(0);
    }
    __syncthreads();  // stage ks visible to all waves

    const unsigned char* buf = &smem[p][0];
    v16bf a[2][3];
    v16bf b[2][3];
    #pragma unroll
    for (int asub = 0; asub < 2; ++asub) {
      const int rbyte = (rowA0 + asub * 16) * ROW_BYTES + aoff;
      #pragma unroll
      for (int m = 0; m < 3; ++m)
        a[asub][m] = load16(buf + m * A_MAT_BYTES + rbyte,
                            buf + m * A_MAT_BYTES + rbyte + 32);
    }
    #pragma unroll
    for (int ns = 0; ns < 2; ++ns) {
      const int cbyte = (colB0 + ns * 16) * ROW_BYTES + boff;
      #pragma unroll
      for (int m = 0; m < 3; ++m)
        b[ns][m] = load16(buf + 3 * A_MAT_BYTES + m * B_MAT_BYTES + cbyte,
                          buf + 3 * A_MAT_BYTES + m * B_MAT_BYTES + cbyte + 32);
    }

    #pragma unroll
    for (int asub = 0; asub < 2; ++asub) {
      #pragma unroll
      for (int ns = 0; ns < 2; ++ns) {
        WMMA_BF16(acc[asub][ns][0], a[asub][0], b[ns][0]);  // mb . mb
        WMMA_BF16(acc[asub][ns][1], a[asub][1], b[ns][0]);  // mh . mb
        WMMA_BF16(acc[asub][ns][2], a[asub][0], b[ns][1]);  // mb . mh
        WMMA_BF16(acc[asub][ns][3], a[asub][2], b[ns][0]);  // mh2. mb
        WMMA_BF16(acc[asub][ns][4], a[asub][0], b[ns][2]);  // mb . mh2
        WMMA_BF16(acc[asub][ns][5], a[asub][1], b[ns][1]);  // mh . mh
      }
    }

    __syncthreads();  // all waves done reading buf[p]
    if (tid < 32 && ks + 2 < NSTAGES)
      tdm_stage(Pmb, Pmh, Pmh2, ibase, jbase, (ks + 2) * KSTEP,
                ldsbase + (unsigned)p * STAGE_BYTES);
  }

  // Epilogue: lane<16 holds rows m=0..7 (VGPR v=m), lane>=16 rows 8..15
  const int mbase = half * 8;
  #pragma unroll
  for (int asub = 0; asub < 2; ++asub) {
    #pragma unroll
    for (int v = 0; v < 8; ++v) {
      const int i = ibase + wr * 32 + asub * 16 + mbase + v;
      const int ami = am[i];
      unsigned bestk = 0xFFFFFFFFu;
      unsigned bestj = 0xFFFFFFFFu;
      #pragma unroll
      for (int ns = 0; ns < 2; ++ns) {
        const int j = jbase + wc * 32 + ns * 16 + l15;
        const float n  = acc[asub][ns][0][v];
        const float s1 = acc[asub][ns][1][v] - acc[asub][ns][2][v];
        const float s2 = acc[asub][ns][3][v] - 2.0f * acc[asub][ns][5][v] + acc[asub][ns][4][v];
        const float nsafe = fmaxf(n, 2.0f);
        const float var = (s2 - s1 * s1 / nsafe) / (nsafe - 1.0f);
        const bool valid = (ami != am[j]) && (i != j) && (n > 1.5f);
        const float score = valid ? var : SENTINEL_F;
        unsigned u = __float_as_uint(score);
        u = (u & 0x80000000u) ? ~u : (u | 0x80000000u);  // total-order encode
        if (u < bestk || (u == bestk && (unsigned)j < bestj)) { bestk = u; bestj = (unsigned)j; }
      }
      #pragma unroll
      for (int m = 1; m < 16; m <<= 1) {
        const unsigned ok = __shfl_xor(bestk, m, 32);
        const unsigned oj = __shfl_xor(bestj, m, 32);
        if (ok < bestk || (ok == bestk && oj < bestj)) { bestk = ok; bestj = oj; }
      }
      if (l15 == 0) {
        const unsigned long long key =
            ((unsigned long long)bestk << 32) | (unsigned long long)bestj;
        atomicMin(&minkeys[i], key);
      }
    }
  }
}

// ---------------- K3a: per-row ||H[i] - H[j*]|| ----------------
__global__ __launch_bounds__(256) void k_rownorm(
    const float* __restrict__ H, const unsigned long long* __restrict__ minkeys,
    float* __restrict__ rownorm) {
  const int row = blockIdx.x;
  const int tid = threadIdx.x;
  const int j = (int)(minkeys[row] & 0xFFFFFFFFull);
  const float* a = H + (size_t)row * D_DIM;
  const float* b = H + (size_t)j * D_DIM;
  float acc = 0.0f;
  for (int d = tid; d < D_DIM; d += 256) {
    const float df = a[d] - b[d];
    acc += df * df;
  }
  __shared__ float ss[256];
  ss[tid] = acc;
  __syncthreads();
  for (int s = 128; s > 0; s >>= 1) {
    if (tid < s) ss[tid] += ss[tid + s];
    __syncthreads();
  }
  if (tid == 0) rownorm[row] = sqrtf(ss[0]);
}

// ---------------- K3b: final deterministic reduction ----------------
__global__ __launch_bounds__(256) void k_final(
    const float* __restrict__ row_mse, const float* __restrict__ rownorm,
    float* __restrict__ out) {
  const int tid = threadIdx.x;
  float acc = 0.0f;
  for (int i = tid; i < T_DIM; i += 256)
    acc += row_mse[i] + ROW_PENALTY * rownorm[i];
  __shared__ float ss[256];
  ss[tid] = acc;
  __syncthreads();
  for (int s = 128; s > 0; s >>= 1) {
    if (tid < s) ss[tid] += ss[tid + s];
    __syncthreads();
  }
  if (tid == 0) out[0] = ss[0];
}

extern "C" void kernel_launch(void* const* d_in, const int* in_sizes, int n_in,
                              void* d_out, int out_size, void* d_ws, size_t ws_size,
                              hipStream_t stream) {
  (void)in_sizes; (void)n_in; (void)out_size; (void)ws_size;
  const float* X = (const float*)d_in[0];
  const float* H = (const float*)d_in[1];
  const float* C = (const float*)d_in[2];
  const float* M = (const float*)d_in[3];
  // d_in[4] is the Python scalar T (==4096), compile-time constant here.

  char* ws = (char*)d_ws;
  const size_t matB = (size_t)T_DIM * D_DIM * sizeof(__bf16);  // 8 MB each
  __bf16* Pmb  = (__bf16*)(ws);
  __bf16* Pmh  = (__bf16*)(ws + matB);
  __bf16* Pmh2 = (__bf16*)(ws + 2 * matB);
  unsigned long long* minkeys = (unsigned long long*)(ws + 3 * matB);
  int*   am      = (int*)  (ws + 3 * matB + (size_t)T_DIM * 8);
  float* row_mse = (float*)(ws + 3 * matB + (size_t)T_DIM * 8 + (size_t)T_DIM * 4);
  float* rownorm = (float*)(ws + 3 * matB + (size_t)T_DIM * 8 + (size_t)T_DIM * 8);
  float* out = (float*)d_out;

  k_init<<<T_DIM / 256, 256, 0, stream>>>(minkeys);
  k_pack<<<T_DIM, 256, 0, stream>>>(X, H, C, M, Pmb, Pmh, Pmh2, am, row_mse);
  dim3 g2(T_DIM / 128, T_DIM / 64);
  k_pair<<<g2, 256, 0, stream>>>(Pmb, Pmh, Pmh2, am, minkeys);
  k_rownorm<<<T_DIM, 256, 0, stream>>>(H, minkeys, rownorm);
  k_final<<<1, 256, 0, stream>>>(row_mse, rownorm, out);
}
